// EmbeddingTransformer_35802847379636
// MI455X (gfx1250) — matrix operand order repaired and checked by reading, not compiled, not run on hardware
//
#include <hip/hip_runtime.h>

#define DEV static __device__ __forceinline__

typedef unsigned short u16;
typedef unsigned int   u32;

typedef __bf16 v16bf __attribute__((ext_vector_type(16)));
typedef float  v8f   __attribute__((ext_vector_type(8)));
typedef int    v4i   __attribute__((ext_vector_type(4)));

// ---- CDNA5 async-to-LDS path (guarded; falls back to plain copies) ----
#if defined(__has_builtin)
# if __has_builtin(__builtin_amdgcn_global_load_async_to_lds_b128)
#  define HAS_ASYNC 1
# endif
#endif
#ifndef HAS_ASYNC
# define HAS_ASYNC 0
#endif

#define N_NEW 8192
#define N_EX  32768
#define DIMK  512
#define NCHUNK 16
#define CPCH   (N_EX / NCHUNK)   // 2048 columns per chunk
#define TILES  (CPCH / 16)       // 128 16-col tiles per chunk
#define NSLOT  (NCHUNK * 2)      // 2 half-wave slots per chunk
#define EPS 1e-8f

// 16-byte chunk type (trivially copyable; avoids HIP vector-type union issues)
struct __attribute__((aligned(16))) Q16 { u32 a, b, c, d; };
struct Q32 { Q16 lo, hi; };
struct __attribute__((aligned(8)))  H4 { u16 a, b, c, d; };

DEV u16 bf_hi(float x){ u32 u = __builtin_bit_cast(u32, x); return (u16)((u + 0x7fffu + ((u >> 16) & 1u)) >> 16); }
DEV float bf_f(u16 h){ u32 u = ((u32)h) << 16; return __builtin_bit_cast(float, u); }

DEV v16bf mk16(Q16 x, Q16 y){ Q32 t{x, y}; return __builtin_bit_cast(v16bf, t); }

// D += A_hi*B_hi + A_hi*B_lo + A_lo*B_hi  (split-bf16 ~ fp32 precision)
DEV v8f wmma_step(v16bf ah, v16bf al, v16bf bh, v16bf bl, v8f c){
  c = __builtin_amdgcn_wmma_f32_16x16x32_bf16(false, ah, false, bh, (short)0, c, false, false);
  c = __builtin_amdgcn_wmma_f32_16x16x32_bf16(false, ah, false, bl, (short)0, c, false, false);
  c = __builtin_amdgcn_wmma_f32_16x16x32_bf16(false, al, false, bh, (short)0, c, false, false);
  return c;
}

#if HAS_ASYNC
typedef __attribute__((address_space(1))) v4i* gv4i_p;  // global
typedef __attribute__((address_space(3))) v4i* lv4i_p;  // LDS
DEV void cp16_async(const u16* g, u16* lds){
  u16* gnc = const_cast<u16*>(g);
  __builtin_amdgcn_global_load_async_to_lds_b128((gv4i_p)gnc, (lv4i_p)lds, 0, 0);
}
DEV void wait_async0(){
# if __has_builtin(__builtin_amdgcn_s_wait_asynccnt)
  __builtin_amdgcn_s_wait_asynccnt(0);
# else
  asm volatile("s_wait_asynccnt 0" ::: "memory");
# endif
}
#else
DEV void wait_async0(){}
#endif

// stage one 16-row x 512-col bf16 tile (16 KB) into LDS with 128 threads
DEV void stage16k(const u16* __restrict__ g, u16* lds, int tid){
  #pragma unroll
  for (int j = 0; j < 8; ++j){
    int lin = tid + j * 128;
#if HAS_ASYNC
    cp16_async(g + lin * 8, lds + lin * 8);
#else
    ((Q16*)lds)[lin] = ((const Q16*)g)[lin];
#endif
  }
}

DEV void top3(float v, int idx, float& v0, int& i0, float& v1, int& i1, float& v2, int& i2){
  if (v > v2){
    if (v > v1){
      v2 = v1; i2 = i1;
      if (v > v0){ v1 = v0; i1 = i0; v0 = v; i0 = idx; }
      else       { v1 = v;  i1 = idx; }
    } else { v2 = v; i2 = idx; }
  }
}

// ---------------- kernel 1: fp32 -> bf16 hi/lo split (+ row L2 norms) ----------------
__global__ __launch_bounds__(128) void k_prep(const float* __restrict__ src,
                                              u16* __restrict__ hi, u16* __restrict__ lo,
                                              float* __restrict__ norm){
  const int row = blockIdx.x, tid = threadIdx.x;
  const float4 x = ((const float4*)(src + (size_t)row * DIMK))[tid];
  const float xs[4] = { x.x, x.y, x.z, x.w };
  u16 h4[4], l4[4];
  float ss = 0.f;
  #pragma unroll
  for (int i = 0; i < 4; ++i){
    const float v = xs[i];
    ss += v * v;
    const u16 h = bf_hi(v);
    h4[i] = h;
    l4[i] = bf_hi(v - bf_f(h));
  }
  ((H4*)(hi + (size_t)row * DIMK))[tid] = H4{h4[0], h4[1], h4[2], h4[3]};
  ((H4*)(lo + (size_t)row * DIMK))[tid] = H4{l4[0], l4[1], l4[2], l4[3]};
  if (norm){
    __shared__ float red[128];
    red[tid] = ss; __syncthreads();
    for (int s = 64; s > 0; s >>= 1){ if (tid < s) red[tid] += red[tid + s]; __syncthreads(); }
    if (tid == 0) norm[row] = sqrtf(red[0]);
  }
}

// ---------------- kernel 2: cosine-sim WMMA + streaming per-row top-3 ----------------
__global__ __launch_bounds__(128) void k_sim_topk(
    const u16* __restrict__ nh, const u16* __restrict__ nl,
    const u16* __restrict__ eh, const u16* __restrict__ el,
    const float* __restrict__ nn, const float* __restrict__ ne,
    float* __restrict__ pv, int* __restrict__ pi){
  __shared__ u16 Ah[64 * DIMK];        // 64 KB
  __shared__ u16 Al[64 * DIMK];        // 64 KB
  __shared__ u16 Bh[2 * 16 * DIMK];    // 32 KB (double buffered)
  __shared__ u16 Bl[2 * 16 * DIMK];    // 32 KB
  __shared__ float Stage[4 * 16 * 16]; // 4 KB sim staging

  const int tid = threadIdx.x;
  const int w = tid >> 5, lane = tid & 31, hf = lane >> 4, l16 = lane & 15;
  const int rbase = blockIdx.x * 64;
  const int chunk = blockIdx.y;
  const int cbase = chunk * CPCH;

  { // stage this block's 64 new-rows (hi+lo) into LDS
    const Q16* gh = (const Q16*)(nh + (size_t)rbase * DIMK);
    const Q16* gl = (const Q16*)(nl + (size_t)rbase * DIMK);
    #pragma unroll
    for (int j = 0; j < 32; ++j){
      const int lin = tid + j * 128;
      ((Q16*)Ah)[lin] = gh[lin];
      ((Q16*)Al)[lin] = gl[lin];
    }
  }

  float nn8[8];
  #pragma unroll
  for (int i = 0; i < 8; ++i) nn8[i] = nn[rbase + w * 16 + hf * 8 + i];

  // prefetch first B tile into buffer 0
  stage16k(eh + (size_t)cbase * DIMK, Bh, tid);
  stage16k(el + (size_t)cbase * DIMK, Bl, tid);

  float v0 = -3e38f, v1 = -3e38f, v2 = -3e38f;
  int   i0 = 0, i1 = 0, i2 = 0;

  const int arow = (w * 16 + l16) * 64 + hf * 2;  // Q16 index base into A (contiguous k = 16*hf + s)
  const int brow = l16 * 64 + hf * 2;         // Q16 index base into B tile

  for (int it = 0; it < TILES; ++it){
    const int buf = it & 1;
    wait_async0();
    __syncthreads();                           // tile(it) + (it==0: A) visible to all waves
    if (it + 1 < TILES){
      const int c1 = cbase + (it + 1) * 16;
      stage16k(eh + (size_t)c1 * DIMK, Bh + (buf ^ 1) * 16 * DIMK, tid);
      stage16k(el + (size_t)c1 * DIMK, Bl + (buf ^ 1) * 16 * DIMK, tid);
    }
    v8f acc = {};
    const Q16* A4h = (const Q16*)Ah;
    const Q16* A4l = (const Q16*)Al;
    const Q16* B4h = (const Q16*)Bh + buf * 1024;
    const Q16* B4l = (const Q16*)Bl + buf * 1024;
    #pragma unroll
    for (int ks = 0; ks < 16; ++ks){
      const int a1 = arow + ks * 4, a2 = a1 + 1;
      const int b1 = brow + ks * 4;
      const v16bf fah = mk16(A4h[a1], A4h[a2]);
      const v16bf fal = mk16(A4l[a1], A4l[a2]);
      const v16bf fbh = mk16(B4h[b1], B4h[b1 + 1]);
      const v16bf fbl = mk16(B4l[b1], B4l[b1 + 1]);
      acc = wmma_step(fah, fal, fbh, fbl, acc);
    }
    const int c0 = cbase + it * 16;
    const float nev = ne[c0 + l16];
    #pragma unroll
    for (int i = 0; i < 8; ++i){
      const float s = acc[i] / (nn8[i] * nev + EPS);   // C layout: M=i+hf*8, N=l16
      Stage[w * 256 + (hf * 8 + i) * 16 + l16] = s;
    }
    __syncthreads();
    #pragma unroll
    for (int j = 0; j < 8; ++j){                       // lane (hf,l16) scans row l16, cols hf*8+j
      const int n = hf * 8 + j;
      top3(Stage[w * 256 + l16 * 16 + n], c0 + n, v0, i0, v1, i1, v2, i2);
    }
  }
  const int row  = rbase + w * 16 + l16;
  const int slot = chunk * 2 + hf;
  const size_t b = ((size_t)row * NSLOT + slot) * 3;
  pv[b] = v0; pv[b + 1] = v1; pv[b + 2] = v2;
  pi[b] = i0; pi[b + 1] = i1; pi[b + 2] = i2;
}

// ---------------- kernel 3: merge partial top-3 + weighted aggregation -> bf16 hi/lo ----------------
__global__ __launch_bounds__(128) void k_merge(
    const float* __restrict__ pv, const int* __restrict__ pi,
    const float* __restrict__ ex,
    u16* __restrict__ ah, u16* __restrict__ al){
  const int row = blockIdx.x, tid = threadIdx.x;
  __shared__ float sv[3]; __shared__ int si[3];
  if (tid == 0){
    float v0 = -3e38f, v1 = -3e38f, v2 = -3e38f; int i0 = 0, i1 = 0, i2 = 0;
    const size_t b = (size_t)row * NSLOT * 3;
    for (int m = 0; m < NSLOT * 3; ++m) top3(pv[b + m], pi[b + m], v0, i0, v1, i1, v2, i2);
    sv[0] = v0; sv[1] = v1; sv[2] = v2; si[0] = i0; si[1] = i1; si[2] = i2;
  }
  __syncthreads();
  const float w0 = sv[0], w1 = sv[1], w2 = sv[2];
  const float inv = 1.f / (w0 + w1 + w2);
  const float4 x0 = ((const float4*)(ex + (size_t)si[0] * DIMK))[tid];
  const float4 x1 = ((const float4*)(ex + (size_t)si[1] * DIMK))[tid];
  const float4 x2 = ((const float4*)(ex + (size_t)si[2] * DIMK))[tid];
  const float g[4] = { (w0 * x0.x + w1 * x1.x + w2 * x2.x) * inv,
                       (w0 * x0.y + w1 * x1.y + w2 * x2.y) * inv,
                       (w0 * x0.z + w1 * x1.z + w2 * x2.z) * inv,
                       (w0 * x0.w + w1 * x1.w + w2 * x2.w) * inv };
  u16 h4[4], l4[4];
  #pragma unroll
  for (int i = 0; i < 4; ++i){ const u16 h = bf_hi(g[i]); h4[i] = h; l4[i] = bf_hi(g[i] - bf_f(h)); }
  ((H4*)(ah + (size_t)row * DIMK))[tid] = H4{h4[0], h4[1], h4[2], h4[3]};
  ((H4*)(al + (size_t)row * DIMK))[tid] = H4{l4[0], l4[1], l4[2], l4[3]};
}

// ---------------- kernel 4: split-bf16 WMMA GEMM  out = X @ W^T + b ----------------
__global__ __launch_bounds__(128) void k_gemm(
    const u16* __restrict__ xh, const u16* __restrict__ xl,
    const u16* __restrict__ wh, const u16* __restrict__ wl,
    const float* __restrict__ bias,
    float* __restrict__ of, u16* __restrict__ oh, u16* __restrict__ ol){
  const int tid = threadIdx.x;
  const int w = tid >> 5, lane = tid & 31, hf = lane >> 4, l16 = lane & 15;
  const int rbase = blockIdx.x * 64 + w * 16;
  const int cbase = blockIdx.y * 16;
  const Q16* X4h = (const Q16*)(xh + (size_t)(rbase + l16) * DIMK);
  const Q16* X4l = (const Q16*)(xl + (size_t)(rbase + l16) * DIMK);
  const Q16* W4h = (const Q16*)(wh + (size_t)(cbase + l16) * DIMK);
  const Q16* W4l = (const Q16*)(wl + (size_t)(cbase + l16) * DIMK);
  v8f acc = {};
  #pragma unroll
  for (int ks = 0; ks < 16; ++ks){
    const int a1 = ks * 4 + hf * 2, a2 = a1 + 1;
    const int b1 = ks * 4 + hf * 2;
    const v16bf fah = mk16(X4h[a1], X4h[a2]);
    const v16bf fal = mk16(X4l[a1], X4l[a2]);
    const v16bf fbh = mk16(W4h[b1], W4h[b1 + 1]);
    const v16bf fbl = mk16(W4l[b1], W4l[b1 + 1]);
    acc = wmma_step(fah, fal, fbh, fbl, acc);
  }
  const float bb = bias[cbase + l16];
  #pragma unroll
  for (int i = 0; i < 8; ++i){
    const int row = rbase + hf * 8 + i;
    const size_t idx = (size_t)row * DIMK + cbase + l16;
    const float val = acc[i] + bb;
    if (of) of[idx] = val;
    if (oh){ const u16 h = bf_hi(val); oh[idx] = h; ol[idx] = bf_hi(val - bf_f(h)); }
  }
}

// ---------------- host ----------------
extern "C" void kernel_launch(void* const* d_in, const int* in_sizes, int n_in,
                              void* d_out, int out_size, void* d_ws, size_t ws_size,
                              hipStream_t stream){
  (void)in_sizes; (void)n_in; (void)out_size; (void)ws_size;
  const float* newf = (const float*)d_in[0];
  const float* exf  = (const float*)d_in[1];
  const float* fcw  = (const float*)d_in[2];
  const float* fcb  = (const float*)d_in[3];
  const float* omw  = (const float*)d_in[4];
  const float* omb  = (const float*)d_in[5];
  float* out = (float*)d_out;

  char* p = (char*)d_ws;
  auto carve = [&](size_t bytes) -> void* {
    void* r = (void*)p; p += (bytes + 255) & ~(size_t)255; return r;
  };
  u16* nh  = (u16*)carve((size_t)N_NEW * DIMK * 2);
  u16* nl  = (u16*)carve((size_t)N_NEW * DIMK * 2);
  u16* eh  = (u16*)carve((size_t)N_EX  * DIMK * 2);
  u16* el  = (u16*)carve((size_t)N_EX  * DIMK * 2);
  u16* w1h = (u16*)carve((size_t)DIMK * DIMK * 2);
  u16* w1l = (u16*)carve((size_t)DIMK * DIMK * 2);
  u16* w2h = (u16*)carve((size_t)DIMK * DIMK * 2);
  u16* w2l = (u16*)carve((size_t)DIMK * DIMK * 2);
  float* nn = (float*)carve((size_t)N_NEW * 4);
  float* ne = (float*)carve((size_t)N_EX * 4);
  float* pv = (float*)carve((size_t)N_NEW * NSLOT * 3 * 4);
  int*   pi = (int*)  carve((size_t)N_NEW * NSLOT * 3 * 4);
  u16* agh = (u16*)carve((size_t)N_NEW * DIMK * 2);
  u16* agl = (u16*)carve((size_t)N_NEW * DIMK * 2);
  u16* hh  = (u16*)carve((size_t)N_NEW * DIMK * 2);
  u16* hl  = (u16*)carve((size_t)N_NEW * DIMK * 2);

  k_prep<<<N_NEW, 128, 0, stream>>>(newf, nh, nl, nn);
  k_prep<<<N_EX,  128, 0, stream>>>(exf,  eh, el, ne);
  k_prep<<<DIMK,  128, 0, stream>>>(fcw,  w1h, w1l, nullptr);
  k_prep<<<DIMK,  128, 0, stream>>>(omw,  w2h, w2l, nullptr);
  k_sim_topk<<<dim3(N_NEW / 64, NCHUNK), 128, 0, stream>>>(nh, nl, eh, el, nn, ne, pv, pi);
  k_merge<<<N_NEW, 128, 0, stream>>>(pv, pi, exf, agh, agl);
  k_gemm<<<dim3(N_NEW / 64, DIMK / 16), 128, 0, stream>>>(agh, agl, w1h, w1l, fcb, nullptr, hh, hl);
  k_gemm<<<dim3(N_NEW / 64, DIMK / 16), 128, 0, stream>>>(hh, hl, w2h, w2l, omb, out, nullptr, nullptr);
}
